// Attention_47940424958078
// MI455X (gfx1250) — compile-verified, matrix operand order checked by reading, not run on hardware
//
#include <hip/hip_runtime.h>

// ---------------------------------------------------------------------------
// Transformer-XL relative attention for MI455X (gfx1250, wave32, WMMA).
// bf16 operands, fp32 accumulation via v_wmma_f32_16x16x32_bf16.
// GEMMs use double-buffered LDS tiles filled with GLOBAL_LOAD_ASYNC_TO_LDS
// (ASYNCcnt) when the toolchain exposes the builtin, else a register-pipelined
// copy.
// ---------------------------------------------------------------------------

typedef __bf16 bf16_t;
typedef __attribute__((ext_vector_type(16))) __bf16 v16bf;
typedef __attribute__((ext_vector_type(8)))  float  v8f;

#define WMMA_BF16(A_, B_, C_) \
  __builtin_amdgcn_wmma_f32_16x16x32_bf16(false, (A_), false, (B_), (short)0, (C_), false, false)

#define D_MODEL 1024
#define N_HEAD  16
#define D_HEAD  64
#define BATCH   2
#define QL      2048
#define BQ      (BATCH * QL)          // 4096 rows of x flattened

// CDNA5 async global->LDS path (probe-guarded; falls back to reg pipeline).
#if defined(__gfx1250__) && defined(__has_builtin)
#if __has_builtin(__builtin_amdgcn_global_load_async_to_lds_b128) && \
    __has_builtin(__builtin_amdgcn_s_wait_asynccnt)
#define HAVE_ASYNC_LDS 1
#endif
#endif
#ifndef HAVE_ASYNC_LDS
#define HAVE_ASYNC_LDS 0
#endif

#if HAVE_ASYNC_LDS
// Parameter type per hipcc diagnostic: pointer to int __attribute__((vector_size(16)))
// in the device (AS1) address space; LDS side in AS3.
typedef __attribute__((__vector_size__(4 * sizeof(int)))) int v4i_t;
typedef __attribute__((address_space(1))) v4i_t* gv4p;
typedef __attribute__((address_space(3))) v4i_t* lv4p;
#endif

// ---------------------------------------------------------------------------
// Fragment loader: 16x32 (MxK) 16-bit A-matrix layout per CDNA5 ISA 7.12.2:
//   lanes 0-15  : row = lane,    K chunks [0..7] and [16..23]
//   lanes 16-31 : row = lane-16, K chunks [8..15] and [24..31]
// B operands are stored transposed (N x K row-major) so the same loader works
// (lane <-> B column); A/B K-chunk pairing is self-consistent.
// ---------------------------------------------------------------------------
__device__ __forceinline__ v16bf frag_ld(const bf16_t* p, int ld) {
  const int lane = threadIdx.x & 31;
  const bf16_t* src = p + (size_t)(lane & 15) * ld + ((lane >> 4) << 3);
  v16bf f;
  bf16_t* fp = reinterpret_cast<bf16_t*>(&f);
  *reinterpret_cast<uint4*>(fp)     = *reinterpret_cast<const uint4*>(src);
  *reinterpret_cast<uint4*>(fp + 8) = *reinterpret_cast<const uint4*>(src + 16);
  return f;
}

// Gathered r_k fragment for the banded rel-shift: B column u <-> r_k row
// t = tbase + (lane&15); rows outside [0, QL) contribute zero (matches the
// rel-shift zero pad / out-of-window semantics).
__device__ __forceinline__ v16bf frag_ld_rk(const bf16_t* rk, int tbase, int kb0) {
  const int lane = threadIdx.x & 31;
  const int t  = tbase + (lane & 15);
  const int kb = kb0 + ((lane >> 4) << 3);
  v16bf f;
  bf16_t* fp = reinterpret_cast<bf16_t*>(&f);
  if (t >= 0 && t < QL) {
    const bf16_t* src = rk + (size_t)t * D_HEAD + kb;
    *reinterpret_cast<uint4*>(fp)     = *reinterpret_cast<const uint4*>(src);
    *reinterpret_cast<uint4*>(fp + 8) = *reinterpret_cast<const uint4*>(src + 16);
  } else {
#pragma unroll
    for (int i = 0; i < 16; ++i) fp[i] = (bf16_t)0.0f;
  }
  return f;
}

// ---------------------------------------------------------------------------
// Elementwise fp32 -> bf16 cast
// ---------------------------------------------------------------------------
__global__ __launch_bounds__(256) void k_cast(const float* __restrict__ src,
                                              bf16_t* __restrict__ dst, int n) {
  int i = blockIdx.x * 256 + threadIdx.x;
  if (i < n) dst[i] = (bf16_t)src[i];
}

// fp32 [R][C] -> bf16 [C][R]  (weights stored N x K for B-fragment loads)
__global__ __launch_bounds__(256) void k_castT(const float* __restrict__ src,
                                               bf16_t* __restrict__ dst, int R, int C) {
  int idx = blockIdx.x * 256 + threadIdx.x;
  int r = idx / C, c = idx - r * C;
  if (r < R) dst[(size_t)c * R + r] = (bf16_t)src[idx];
}

// ---------------------------------------------------------------------------
// Tiled bf16 WMMA GEMM: C[M][N] (fp32) = A[M][K] * Bt[N][K]^T
// 256 threads = 8 waves; 128x128x32 tile; wave = 64x32 (8 accumulators).
// Double-buffered LDS: tile k+1 streams in (async global->LDS) while tile k's
// 8 WMMAs execute. Assumes M%128 == N%128 == K%32 == 0 (true here).
// ---------------------------------------------------------------------------
__global__ __launch_bounds__(256)
void k_gemm_bf16(const bf16_t* __restrict__ A, const bf16_t* __restrict__ Bt,
                 float* __restrict__ C, int M, int N, int K) {
  __shared__ bf16_t As[2][128 * 32];
  __shared__ bf16_t Bs[2][128 * 32];
  const int tid = threadIdx.x, lane = tid & 31, wave = tid >> 5;
  const int wm = wave >> 2, wn = wave & 3;
  const int m0 = blockIdx.y * 128, n0 = blockIdx.x * 128;

  // Each thread moves two 16B chunks per matrix: rows r0 and r0+64, col cb.
  const int r0 = tid >> 2, cb = (tid & 3) << 3;
  const size_t gA0 = (size_t)(m0 + r0) * K + cb;
  const size_t gA1 = (size_t)(m0 + r0 + 64) * K + cb;
  const size_t gB0 = (size_t)(n0 + r0) * K + cb;
  const size_t gB1 = (size_t)(n0 + r0 + 64) * K + cb;
  const int l0 = r0 * 32 + cb, l1 = (r0 + 64) * 32 + cb;

  v8f acc[4][2];
#pragma unroll
  for (int mi = 0; mi < 4; ++mi)
#pragma unroll
    for (int ni = 0; ni < 2; ++ni)
#pragma unroll
      for (int r = 0; r < 8; ++r) acc[mi][ni][r] = 0.f;

#if HAVE_ASYNC_LDS
  auto issue = [&](int p_, int k0_) {
    __builtin_amdgcn_global_load_async_to_lds_b128((gv4p)(A + gA0 + k0_),  (lv4p)&As[p_][l0], 0, 0);
    __builtin_amdgcn_global_load_async_to_lds_b128((gv4p)(A + gA1 + k0_),  (lv4p)&As[p_][l1], 0, 0);
    __builtin_amdgcn_global_load_async_to_lds_b128((gv4p)(Bt + gB0 + k0_), (lv4p)&Bs[p_][l0], 0, 0);
    __builtin_amdgcn_global_load_async_to_lds_b128((gv4p)(Bt + gB1 + k0_), (lv4p)&Bs[p_][l1], 0, 0);
  };
  issue(0, 0);
#else
  uint4 ra0, ra1, rb0, rb1;
  auto fetch = [&](int k0_) {
    ra0 = *reinterpret_cast<const uint4*>(A + gA0 + k0_);
    ra1 = *reinterpret_cast<const uint4*>(A + gA1 + k0_);
    rb0 = *reinterpret_cast<const uint4*>(Bt + gB0 + k0_);
    rb1 = *reinterpret_cast<const uint4*>(Bt + gB1 + k0_);
  };
  fetch(0);
#endif

  int p = 0;
  for (int k0 = 0; k0 < K; k0 += 32, p ^= 1) {
    const bool more = (k0 + 32) < K;
#if HAVE_ASYNC_LDS
    if (more) {
      issue(p ^ 1, k0 + 32);                 // next tile in flight during compute
      __builtin_amdgcn_s_wait_asynccnt(4);   // retire only current tile's 4 copies
    } else {
      __builtin_amdgcn_s_wait_asynccnt(0);
    }
#else
    // Drain current tile from registers into LDS buffer p (loadcnt wait here
    // overlapped previous iteration's WMMAs), then launch next tile's loads.
    *reinterpret_cast<uint4*>(&As[p][l0]) = ra0;
    *reinterpret_cast<uint4*>(&As[p][l1]) = ra1;
    *reinterpret_cast<uint4*>(&Bs[p][l0]) = rb0;
    *reinterpret_cast<uint4*>(&Bs[p][l1]) = rb1;
    if (more) fetch(k0 + 32);
#endif
    __syncthreads();

    v16bf af[4];
#pragma unroll
    for (int mi = 0; mi < 4; ++mi) af[mi] = frag_ld(&As[p][(wm * 64 + mi * 16) * 32], 32);
#pragma unroll
    for (int ni = 0; ni < 2; ++ni) {
      v16bf bfrag = frag_ld(&Bs[p][(wn * 32 + ni * 16) * 32], 32);
#pragma unroll
      for (int mi = 0; mi < 4; ++mi)
        acc[mi][ni] = WMMA_BF16(af[mi], bfrag, acc[mi][ni]);
    }
    __syncthreads();
  }

  // C tile layout: lane n -> col n%16; lanes>=16 -> rows +8; vgpr r -> row r.
  const int colh = lane & 15, rowh = (lane >> 4) << 3;
#pragma unroll
  for (int mi = 0; mi < 4; ++mi)
#pragma unroll
    for (int ni = 0; ni < 2; ++ni) {
      int cm = m0 + wm * 64 + mi * 16 + rowh;
      int cn = n0 + wn * 32 + ni * 16 + colh;
#pragma unroll
      for (int r = 0; r < 8; ++r)
        C[(size_t)(cm + r) * N + cn] = acc[mi][ni][r];
    }
}

// ---------------------------------------------------------------------------
// Pack QKV: w_heads fp32 [4096][3072] ->
//   Qw,Qr bf16 [b][h][q][64] (+ r_w_bias / r_r_bias), Kp bf16 [b][h][k][64],
//   Vt bf16 [b][h][64][k] (transposed so P*V B-fragments are contiguous in K).
// ---------------------------------------------------------------------------
__global__ __launch_bounds__(256)
void k_pack_qkv(const float* __restrict__ wh, const float* __restrict__ rwb,
                const float* __restrict__ rrb, bf16_t* __restrict__ Qw,
                bf16_t* __restrict__ Qr, bf16_t* __restrict__ Kp,
                bf16_t* __restrict__ Vt) {
  int idx = blockIdx.x * 256 + threadIdx.x;           // < 4096*3072
  int row = idx / 3072, col = idx - row * 3072;
  int b = row >> 11, i = row & (QL - 1);
  int seg = col >> 10, cc = col & 1023;
  int h = cc >> 6, d = cc & 63;
  float v = wh[idx];
  size_t base = (((size_t)(b * N_HEAD + h) * QL) + i) * D_HEAD + d;
  if (seg == 0) {
    Qw[base] = (bf16_t)(v + rwb[cc]);
    Qr[base] = (bf16_t)(v + rrb[cc]);
  } else if (seg == 1) {
    Kp[base] = (bf16_t)v;
  } else {
    Vt[((size_t)(b * N_HEAD + h) * D_HEAD + d) * QL + i] = (bf16_t)v;
  }
}

// r_k fp32 [2048][1024] -> Rk bf16 [h][2048][64]
__global__ __launch_bounds__(256)
void k_pack_rk(const float* __restrict__ rk, bf16_t* __restrict__ Rk) {
  int idx = blockIdx.x * 256 + threadIdx.x;           // < 2048*1024
  int t = idx >> 10, cc = idx & 1023;
  int h = cc >> 6, d = cc & 63;
  Rk[((size_t)h * QL + t) * D_HEAD + d] = (bf16_t)rk[idx];
}

// ---------------------------------------------------------------------------
// Flash attention with Transformer-XL banded relative shift.
// grid (qtile=32, head=16, batch=2); 128 threads = 4 waves; 16 q-rows/wave.
// Per 32-wide k-tile per wave: 4 AC + 8 BD + 4 PV WMMAs (16x16x32 bf16).
// All LDS use is wave-private -> no block barriers in the k loop
// (DS ops are in-order per wave; compiler inserts s_wait_dscnt).
// ---------------------------------------------------------------------------
__global__ __launch_bounds__(128)
void k_attn(const bf16_t* __restrict__ Qw, const bf16_t* __restrict__ Qr,
            const bf16_t* __restrict__ Kp, const bf16_t* __restrict__ Vt,
            const bf16_t* __restrict__ Rk, const unsigned char* __restrict__ mask,
            bf16_t* __restrict__ ctx) {
  __shared__ float  E_lds[4][16][32];   // banded BD staging (per wave)
  __shared__ bf16_t P_lds[4][16][32];   // C-layout -> A-layout transit (per wave)

  const int lane = threadIdx.x & 31, wave = threadIdx.x >> 5;
  const int qt = blockIdx.x, h = blockIdx.y, b = blockIdx.z;
  const size_t bh  = ((size_t)b * N_HEAD + h) * QL * D_HEAD;
  const size_t vtb = ((size_t)b * N_HEAD + h) * D_HEAD * QL;
  const size_t rkb = (size_t)h * QL * D_HEAD;
  const int mb = b * QL;
  const int i0 = qt * 64 + wave * 16;
  const int colh = lane & 15, rowh = (lane >> 4) << 3;
  const float NEGINF = -__builtin_inff();

  v16bf Aw[2], Ar[2];
#pragma unroll
  for (int ks = 0; ks < 2; ++ks) {
    Aw[ks] = frag_ld(Qw + bh + (size_t)i0 * D_HEAD + ks * 32, D_HEAD);
    Ar[ks] = frag_ld(Qr + bh + (size_t)i0 * D_HEAD + ks * 32, D_HEAD);
  }

  v8f m_v, l_v, O[4];
#pragma unroll
  for (int r = 0; r < 8; ++r) { m_v[r] = NEGINF; l_v[r] = 0.f; }
#pragma unroll
  for (int t = 0; t < 4; ++t)
#pragma unroll
    for (int r = 0; r < 8; ++r) O[t][r] = 0.f;

  for (int j0 = 0; j0 < QL; j0 += 32) {
    // L2 prefetch of next K/V tiles (gfx1250 global_prefetch).
    if (j0 + 32 < QL) {
      __builtin_prefetch(Kp + bh + (size_t)(j0 + 32 + lane) * D_HEAD, 0, 0);
      __builtin_prefetch(Vt + vtb + (size_t)(2 * lane) * QL + j0 + 32, 0, 0);
    }

    v8f S[2];
#pragma unroll
    for (int ni = 0; ni < 2; ++ni)
#pragma unroll
      for (int r = 0; r < 8; ++r) S[ni][r] = 0.f;

    // ---- content scores: S += (w_q + r_w_bias) . K^T ----
#pragma unroll
    for (int ni = 0; ni < 2; ++ni)
#pragma unroll
      for (int ks = 0; ks < 2; ++ks) {
        v16bf kf = frag_ld(Kp + bh + (size_t)(j0 + ni * 16) * D_HEAD + ks * 32, D_HEAD);
        S[ni] = WMMA_BF16(Aw[ks], kf, S[ni]);
      }

    // ---- position scores: banded rel-shift via E[p,u] = q_r . r_k[t0-15+u] ----
#pragma unroll
    for (int ni = 0; ni < 2; ++ni) {
      const int t0 = j0 + ni * 16 - i0 + (QL - 1);
      v8f E[2];
#pragma unroll
      for (int ut = 0; ut < 2; ++ut)
#pragma unroll
        for (int r = 0; r < 8; ++r) E[ut][r] = 0.f;
#pragma unroll
      for (int ut = 0; ut < 2; ++ut)
#pragma unroll
        for (int ks = 0; ks < 2; ++ks) {
          v16bf rf = frag_ld_rk(Rk + rkb, t0 - 15 + ut * 16, ks * 32);
          E[ut] = WMMA_BF16(Ar[ks], rf, E[ut]);
        }
#pragma unroll
      for (int ut = 0; ut < 2; ++ut)
#pragma unroll
        for (int r = 0; r < 8; ++r)
          E_lds[wave][rowh + r][ut * 16 + colh] = E[ut][r];
#pragma unroll
      for (int r = 0; r < 8; ++r) {
        int p = rowh + r;
        S[ni][r] += E_lds[wave][p][colh - p + 15];   // bd[p,c] = E[p, c-p+15]
      }
    }

    // ---- scale + key mask ----
#pragma unroll
    for (int ni = 0; ni < 2; ++ni) {
      const int j = j0 + ni * 16 + colh;
      const float msk = mask[mb + j] ? NEGINF : 0.f;
#pragma unroll
      for (int r = 0; r < 8; ++r) S[ni][r] = S[ni][r] * 0.125f + msk;
    }

    // ---- online softmax: row reductions across the 16-lane half-wave ----
    v8f alpha;
#pragma unroll
    for (int r = 0; r < 8; ++r) {
      float v = fmaxf(S[0][r], S[1][r]);
#pragma unroll
      for (int md = 1; md <= 8; md <<= 1) v = fmaxf(v, __shfl_xor(v, md, 32));
      float mn = fmaxf(m_v[r], v);
      alpha[r] = (mn == NEGINF) ? 1.f : __expf(m_v[r] - mn);
      m_v[r] = mn;
    }
#pragma unroll
    for (int r = 0; r < 8; ++r) {
      float mn = m_v[r];
      float p0 = (S[0][r] == NEGINF) ? 0.f : __expf(S[0][r] - mn);
      float p1 = (S[1][r] == NEGINF) ? 0.f : __expf(S[1][r] - mn);
      S[0][r] = p0; S[1][r] = p1;
      float rs = p0 + p1;
#pragma unroll
      for (int md = 1; md <= 8; md <<= 1) rs += __shfl_xor(rs, md, 32);
      l_v[r] = l_v[r] * alpha[r] + rs;
    }
#pragma unroll
    for (int t = 0; t < 4; ++t)
#pragma unroll
      for (int r = 0; r < 8; ++r) O[t][r] *= alpha[r];

    // ---- P (C-layout) -> LDS -> A-layout fragment; accumulate P.V ----
#pragma unroll
    for (int ni = 0; ni < 2; ++ni)
#pragma unroll
      for (int r = 0; r < 8; ++r)
        P_lds[wave][rowh + r][ni * 16 + colh] = (bf16_t)S[ni][r];
    v16bf pf = frag_ld(&P_lds[wave][0][0], 32);
#pragma unroll
    for (int t = 0; t < 4; ++t) {
      v16bf vf = frag_ld(Vt + vtb + (size_t)(t * 16) * QL + j0, QL);
      O[t] = WMMA_BF16(pf, vf, O[t]);
    }
  }

  // ---- epilogue: O /= l; write bf16 context [b*q][h*64+d] ----
#pragma unroll
  for (int r = 0; r < 8; ++r) {
    float inv = (l_v[r] > 0.f) ? 1.f / l_v[r] : 0.f;
#pragma unroll
    for (int t = 0; t < 4; ++t) O[t][r] *= inv;
  }
  const int gi = b * QL + i0 + rowh;
#pragma unroll
  for (int t = 0; t < 4; ++t)
#pragma unroll
    for (int r = 0; r < 8; ++r)
      ctx[(size_t)(gi + r) * D_MODEL + h * D_HEAD + t * 16 + colh] = (bf16_t)O[t][r];
}

// ---------------------------------------------------------------------------
// Fused residual + LayerNorm: one 256-thread block per row of 1024.
// ---------------------------------------------------------------------------
__global__ __launch_bounds__(256)
void k_ln(const float* __restrict__ x, const float* __restrict__ attn,
          const float* __restrict__ gamma, const float* __restrict__ beta,
          float* __restrict__ out) {
  __shared__ float red[2][8];
  const size_t row = blockIdx.x;
  float s = 0.f, ss = 0.f;
  for (int c = threadIdx.x; c < D_MODEL; c += 256) {
    float v = x[row * D_MODEL + c] + attn[row * D_MODEL + c];
    s += v; ss += v * v;
  }
#pragma unroll
  for (int md = 1; md <= 16; md <<= 1) { s += __shfl_xor(s, md, 32); ss += __shfl_xor(ss, md, 32); }
  if ((threadIdx.x & 31) == 0) { red[0][threadIdx.x >> 5] = s; red[1][threadIdx.x >> 5] = ss; }
  __syncthreads();
  float S = 0.f, SS = 0.f;
#pragma unroll
  for (int w = 0; w < 8; ++w) { S += red[0][w]; SS += red[1][w]; }
  const float mu = S * (1.f / D_MODEL);
  const float var = SS * (1.f / D_MODEL) - mu * mu;
  const float rs = rsqrtf(var + 1e-5f);
  for (int c = threadIdx.x; c < D_MODEL; c += 256) {
    float v = x[row * D_MODEL + c] + attn[row * D_MODEL + c];
    out[row * D_MODEL + c] = (v - mu) * rs * gamma[c] + beta[c];
  }
}

// ---------------------------------------------------------------------------
// Orchestration
// ---------------------------------------------------------------------------
extern "C" void kernel_launch(void* const* d_in, const int* in_sizes, int n_in,
                              void* d_out, int out_size, void* d_ws, size_t ws_size,
                              hipStream_t stream) {
  const float* x      = (const float*)d_in[0];
  const float* relpos = (const float*)d_in[1];
  const float* rwb    = (const float*)d_in[2];
  const float* rrb    = (const float*)d_in[3];
  const unsigned char* mask = (const unsigned char*)d_in[4];   // jnp bool -> 1 byte
  const float* Wqkv   = (const float*)d_in[5];
  const float* Wrel   = (const float*)d_in[6];
  const float* Wout   = (const float*)d_in[7];
  const float* gamma  = (const float*)d_in[8];
  const float* beta   = (const float*)d_in[9];
  float* out = (float*)d_out;

  char* ws = (char*)d_ws;
  size_t off = 0;
  auto alloc = [&](size_t bytes) -> void* {
    off = (off + 255) & ~(size_t)255;
    void* p = ws + off;
    off += bytes;
    return p;
  };

  bf16_t* xb     = (bf16_t*)alloc((size_t)BQ * D_MODEL * 2);          // x bf16
  bf16_t* rpb    = (bf16_t*)alloc((size_t)QL * D_MODEL * 2);          // rel_pos bf16
  bf16_t* WqkvT  = (bf16_t*)alloc((size_t)3 * D_MODEL * D_MODEL * 2); // [3072][1024]
  bf16_t* WrelT  = (bf16_t*)alloc((size_t)D_MODEL * D_MODEL * 2);
  bf16_t* WoutT  = (bf16_t*)alloc((size_t)D_MODEL * D_MODEL * 2);
  float*  wh     = (float*)alloc((size_t)BQ * 3 * D_MODEL * 4);       // w_heads fp32
  float*  rkf    = (float*)alloc((size_t)QL * D_MODEL * 4);           // r_k fp32
  bf16_t* Qw     = (bf16_t*)alloc((size_t)BQ * D_MODEL * 2);
  bf16_t* Qr     = (bf16_t*)alloc((size_t)BQ * D_MODEL * 2);
  bf16_t* Kp     = (bf16_t*)alloc((size_t)BQ * D_MODEL * 2);
  bf16_t* Vt     = (bf16_t*)alloc((size_t)BQ * D_MODEL * 2);
  bf16_t* Rk     = (bf16_t*)alloc((size_t)N_HEAD * QL * D_HEAD * 2);
  bf16_t* ctx    = (bf16_t*)alloc((size_t)BQ * D_MODEL * 2);
  float*  attnO  = (float*)alloc((size_t)BQ * D_MODEL * 4);

  // 1) casts / weight transposes to bf16
  k_cast<<<(BQ * D_MODEL) / 256, 256, 0, stream>>>(x, xb, BQ * D_MODEL);
  k_cast<<<(QL * D_MODEL) / 256, 256, 0, stream>>>(relpos, rpb, QL * D_MODEL);
  k_castT<<<(D_MODEL * 3 * D_MODEL) / 256, 256, 0, stream>>>(Wqkv, WqkvT, D_MODEL, 3 * D_MODEL);
  k_castT<<<(D_MODEL * D_MODEL) / 256, 256, 0, stream>>>(Wrel, WrelT, D_MODEL, D_MODEL);
  k_castT<<<(D_MODEL * D_MODEL) / 256, 256, 0, stream>>>(Wout, WoutT, D_MODEL, D_MODEL);

  // 2) projections: w_heads = x @ W_qkv ; r_k = rel_pos @ W_rel
  k_gemm_bf16<<<dim3((3 * D_MODEL) / 128, BQ / 128), 256, 0, stream>>>(
      xb, WqkvT, wh, BQ, 3 * D_MODEL, D_MODEL);
  k_gemm_bf16<<<dim3(D_MODEL / 128, QL / 128), 256, 0, stream>>>(
      rpb, WrelT, rkf, QL, D_MODEL, D_MODEL);

  // 3) pack per-head Q(+biases)/K/V^T and Rk in bf16
  k_pack_qkv<<<(BQ * 3 * D_MODEL) / 256, 256, 0, stream>>>(wh, rwb, rrb, Qw, Qr, Kp, Vt);
  k_pack_rk<<<(QL * D_MODEL) / 256, 256, 0, stream>>>(rkf, Rk);

  // 4) flash attention with banded relative shift
  k_attn<<<dim3(QL / 64, N_HEAD, BATCH), 128, 0, stream>>>(Qw, Qr, Kp, Vt, Rk, mask, ctx);

  // 5) output projection + fused residual LayerNorm
  k_gemm_bf16<<<dim3(D_MODEL / 128, BQ / 128), 256, 0, stream>>>(
      ctx, WoutT, attnO, BQ, D_MODEL, D_MODEL);
  k_ln<<<BQ, 256, 0, stream>>>(x, attnO, gamma, beta, out);

  (void)in_sizes; (void)n_in; (void)out_size; (void)ws_size;
}
